// RETSim_7344394076302
// MI455X (gfx1250) — matrix operand order repaired and checked by reading, not compiled, not run on hardware
//
#include <hip/hip_runtime.h>
#include <hip/hip_bf16.h>
#include <math.h>

// ---------- constants from the reference ----------
#define BB   128
#define TT   512
#define BIN  24
#define DD   256
#define SS   128
#define LL   2
#define ROWS (BB*TT)            // 65536
#define NUV  (2*DD + SS)        // 640
#define EPSV 1e-6f
#define INV_MAXLEN (1.0f/512.0f)
#define LN10000 9.210340371976184f

// ---------- vector typedefs (union-safe ext vectors, not HIP structs) ----------
typedef __attribute__((ext_vector_type(16))) __bf16        bf16x16;
typedef __attribute__((ext_vector_type(8)))  float         f32x8;
typedef __attribute__((ext_vector_type(4)))  unsigned int  uintx4;
typedef __attribute__((ext_vector_type(4)))  float         f32x4;

union Frag { bf16x16 v; uintx4 u[2]; };

__device__ __forceinline__ float swishf(float x) {
    return x * (1.0f / (1.0f + __expf(-x)));
}

// ============================================================
// fp32 -> bf16 weight conversion
// ============================================================
__global__ void f2bf_kernel(const float* __restrict__ src, __bf16* __restrict__ dst, int n) {
    int i = blockIdx.x * 256 + threadIdx.x;
    if (i < n) dst[i] = (__bf16)src[i];
}

// ============================================================
// sinusoid positional table: pe[t*24 + c]
// ============================================================
__global__ void pe_kernel(float* __restrict__ pe) {
    int idx = blockIdx.x * 256 + threadIdx.x;
    if (idx >= TT * BIN) return;
    int t = idx / BIN, c = idx % BIN;
    int cc = (c < 12) ? c : c - 12;
    float inv = __expf((float)cc * (-LN10000 / 11.0f));   // num=12, num-1=11
    float ang = (float)t * inv;
    pe[idx] = (c < 12) ? __sinf(ang) : __cosf(ang);
}

// ============================================================
// embed: h = swish((x + ps*pe) @ W_in + b_in)   K=24 scalar (trivial FLOPs)
// ============================================================
__global__ void embed_kernel(const float* __restrict__ x, const float* __restrict__ pe,
                             const float* __restrict__ ps_ptr,
                             const float* __restrict__ W_in, const float* __restrict__ b_in,
                             float* __restrict__ h) {
    int idx = blockIdx.x * 256 + threadIdx.x;          // B*T*D threads
    int d  = idx & (DD - 1);
    int bt = idx >> 8;
    int t  = bt & (TT - 1);
    float ps = ps_ptr[0];
    const float* xr  = x  + (size_t)bt * BIN;
    const float* per = pe + (size_t)t  * BIN;
    float acc = b_in[d];
#pragma unroll
    for (int c = 0; c < BIN; ++c)
        acc += (xr[c] + ps * per[c]) * W_in[c * DD + d];
    h[idx] = swishf(acc);
}

// ============================================================
// RMSNorm: one wave per row of 256, bf16 output
// ============================================================
__global__ void rmsnorm_kernel(const float* __restrict__ h, const float* __restrict__ ns,
                               int layer, __bf16* __restrict__ hn) {
    int lane = threadIdx.x & 31, w = threadIdx.x >> 5;
    int row = blockIdx.x * 8 + w;
    const float* hr = h + (size_t)row * DD + lane * 8;
    f32x4 a = *(const f32x4*)hr;
    f32x4 b = *(const f32x4*)(hr + 4);
    float s = a[0]*a[0]+a[1]*a[1]+a[2]*a[2]+a[3]*a[3]
            + b[0]*b[0]+b[1]*b[1]+b[2]*b[2]+b[3]*b[3];
#pragma unroll
    for (int off = 16; off > 0; off >>= 1) s += __shfl_xor(s, off, 32);
    float scale = rsqrtf(s * (1.0f / DD) + EPSV) * ns[layer];
    __bf16* o = hn + (size_t)row * DD + lane * 8;
#pragma unroll
    for (int j = 0; j < 4; ++j) o[j]     = (__bf16)(a[j] * scale);
#pragma unroll
    for (int j = 0; j < 4; ++j) o[4 + j] = (__bf16)(b[j] * scale);
}

// ============================================================
// Generic WMMA GEMM: C = act(A(bf16,MxK) @ B(bf16,KxN) + bias) [+ resid]
// block=256 (8 waves); each wave owns a 16x64 output strip (4 accumulators
// sharing one A fragment per K-step -> 4x A reuse). All loads (2 b128 for A, 64 scalar for B) of a
// K-step are issued before the 4 WMMAs so the scheduler can overlap them.
// grid=(N/64, M/128)
// ============================================================
__global__ void gemm_wmma_kernel(const __bf16* __restrict__ A, int lda,
                                 const __bf16* __restrict__ Bm, int ldb,
                                 const float* __restrict__ bias,
                                 const float* __restrict__ resid,
                                 float* __restrict__ outF, __bf16* __restrict__ outB,
                                 int ldo, int M, int N, int K, int act) {
    int lane = threadIdx.x & 31, w = threadIdx.x >> 5;
    int rowBase = blockIdx.y * 128 + w * 16;
    int colBase = blockIdx.x * 64;
    if (rowBase >= M) return;                         // wave-uniform
    int half = lane >> 4, mloc = lane & 15;

    f32x8 c[4];
#pragma unroll
    for (int nt = 0; nt < 4; ++nt)
#pragma unroll
        for (int i = 0; i < 8; ++i) c[nt][i] = 0.0f;

    const __bf16* arow = A + (size_t)(rowBase + mloc) * lda + 8 * half;
    const __bf16* brow = Bm + colBase + mloc;   // advanced by 32*ldb
    const size_t bstep = (size_t)32 * ldb;

    for (int k0 = 0; k0 < K; k0 += 32) {
        Frag af, bf[4];
        af.u[0] = *(const uintx4*)(arow + k0);        // K = 8h..8h+7
        af.u[1] = *(const uintx4*)(arow + k0 + 16);   // K = 16+8h..
#pragma unroll
        for (int nt = 0; nt < 4; ++nt) {
#pragma unroll
            for (int e = 0; e < 16; ++e) {
                int kk = (e & 7) + 8 * half + 16 * (e >> 3);
                bf[nt].v[e] = brow[(size_t)kk * ldb + nt * 16];
            }
        }
#pragma unroll
        for (int nt = 0; nt < 4; ++nt)
            c[nt] = __builtin_amdgcn_wmma_f32_16x16x32_bf16(false, af.v, false, bf[nt].v,
                                                            (short)0, c[nt], false, false);
        brow += bstep;
    }
#pragma unroll
    for (int nt = 0; nt < 4; ++nt) {
#pragma unroll
        for (int v = 0; v < 8; ++v) {
            int m = v + 8 * half, n = lane & 15;
            int row = rowBase + m, col = colBase + nt * 16 + n;
            float val = c[nt][v];
            if (bias) val += bias[col];
            if (act)  val = swishf(val);
            if (resid) val += resid[(size_t)row * ldo + col];
            if (outF) outF[(size_t)row * ldo + col] = val;
            else      outB[(size_t)row * ldo + col] = (__bf16)val;
        }
    }
}

// ============================================================
// RoPE on base slice of uv -> q (row-major) and kT (dim-major: kT[b,dim,t])
// the attention QK^T B-fragment is read from it by TT-strided scalar loads.
// gamma/beta pre-offset to layer: [0:128]=row0(q), [128:256]=row1(k)
// ============================================================
__global__ void rope_kernel(const __bf16* __restrict__ uv,
                            const float* __restrict__ gamma, const float* __restrict__ beta,
                            __bf16* __restrict__ q, __bf16* __restrict__ kT) {
    int idx = blockIdx.x * 256 + threadIdx.x;   // ROWS*128 threads
    int j   = idx & 127;
    int row = idx >> 7;
    int t   = row & (TT - 1);
    int b   = row >> 9;
    int jm  = j & 63;
    float inv = __expf(-(float)(2 * jm) * (LN10000 / 128.0f));
    float ang = (float)t * inv;
    float cs = __cosf(ang), sn = __sinf(ang);
    int p = j ^ 64;
    float sgn = (j < 64) ? -1.0f : 1.0f;
    const __bf16* base = uv + (size_t)row * NUV + 2 * DD;
    float xj = (float)base[j], xp = (float)base[p];
    float qj = xj * gamma[j] + beta[j];
    float qp = xp * gamma[p] + beta[p];
    q[(size_t)row * SS + j] = (__bf16)(qj * cs + sgn * qp * sn);
    float kj = xj * gamma[SS + j] + beta[SS + j];
    float kp = xp * gamma[SS + p] + beta[SS + p];
    kT[((size_t)b * SS + j) * TT + t] = (__bf16)(kj * cs + sgn * kp * sn);
}

// ============================================================
// Fused attention: per (batch, 16-row query tile):
//   scores = relu(qk^T/512)^2  (WMMA, K=128, B-frags from kT by strided loads)
//   acc    = scores @ V        (WMMA, key tiles paired into K=32)
//   uv[:, 0:256] = u * acc     (in place: y for the W2 GEMM)
// block=128 (4 waves, independent row tiles); LDS 16x32 f32 per wave
// for the C-layout -> A-layout relayout of the score tile.
// ============================================================
__global__ void attn_kernel(const __bf16* __restrict__ q, const __bf16* __restrict__ kT,
                            __bf16* __restrict__ uv) {
    __shared__ float sm[4][16][32];
    int lane = threadIdx.x & 31, w = threadIdx.x >> 5;
    int half = lane >> 4, mloc = lane & 15, n15 = lane & 15;
    int rt = blockIdx.x * 4 + w;       // 0..31 query tiles
    int b  = blockIdx.y;

    const size_t qrow = (size_t)(b * TT + rt * 16 + mloc) * SS + 8 * half;
    Frag aq[4];
#pragma unroll
    for (int ch = 0; ch < 4; ++ch) {
        aq[ch].u[0] = *(const uintx4*)(q + qrow + ch * 32);
        aq[ch].u[1] = *(const uintx4*)(q + qrow + ch * 32 + 16);
    }
    const __bf16* kbase = kT + (size_t)b * SS * TT + n15;  // + ch*32*TT + key

    f32x8 acc[16];
#pragma unroll
    for (int i = 0; i < 16; ++i)
#pragma unroll
        for (int j = 0; j < 8; ++j) acc[i][j] = 0.0f;

    for (int kp = 0; kp < TT / 32; ++kp) {           // key-tile pairs (32 keys)
#pragma unroll
        for (int sub = 0; sub < 2; ++sub) {
            int kt = kp * 2 + sub;
            f32x8 sc;
#pragma unroll
            for (int j = 0; j < 8; ++j) sc[j] = 0.0f;
            Frag bk[4];                              // preload all K=128 chunks
#pragma unroll
            for (int ch = 0; ch < 4; ++ch) {
                const __bf16* krow = kbase + (size_t)(ch * 32) * TT + kt * 16;
#pragma unroll
                for (int e = 0; e < 16; ++e) {
                    int kk = (e & 7) + 8 * half + 16 * (e >> 3);
                    bk[ch].v[e] = krow[(size_t)kk * TT];
                }
            }
#pragma unroll
            for (int ch = 0; ch < 4; ++ch)
                sc = __builtin_amdgcn_wmma_f32_16x16x32_bf16(false, aq[ch].v, false, bk[ch].v,
                                                             (short)0, sc, false, false);
#pragma unroll
            for (int v = 0; v < 8; ++v) {            // relu(s/512)^2, stash in LDS
                float s = sc[v] * INV_MAXLEN;
                s = (s > 0.0f) ? s * s : 0.0f;
                sm[w][v + 8 * half][n15 + sub * 16] = s;
            }
        }
        Frag aA;                                     // re-read as 16x32 A fragment
#pragma unroll
        for (int e = 0; e < 16; ++e) {
            int kk = (e & 7) + 8 * half + 16 * (e >> 3);
            aA.v[e] = (__bf16)sm[w][mloc][kk];
        }
        const __bf16* vb = uv + (size_t)(b * TT + kp * 32) * NUV + DD + n15; // V slice
#pragma unroll
        for (int ng = 0; ng < 4; ++ng) {             // 16x32 @ 32x256, 4 tiles at a time
            Frag bv[4];
#pragma unroll
            for (int j = 0; j < 4; ++j) {
#pragma unroll
                for (int e = 0; e < 16; ++e) {
                    int kk = (e & 7) + 8 * half + 16 * (e >> 3);
                    bv[j].v[e] = vb[(size_t)kk * NUV + (ng * 4 + j) * 16];
                }
            }
#pragma unroll
            for (int j = 0; j < 4; ++j)
                acc[ng * 4 + j] = __builtin_amdgcn_wmma_f32_16x16x32_bf16(
                    false, aA.v, false, bv[j].v, (short)0, acc[ng * 4 + j], false, false);
        }
    }
    // y = u * acc, written over u (each row/col touched by exactly one lane)
#pragma unroll
    for (int nc = 0; nc < 16; ++nc) {
#pragma unroll
        for (int v = 0; v < 8; ++v) {
            int m = v + 8 * half;
            int col = nc * 16 + n15;
            size_t idx = (size_t)(b * TT + rt * 16 + m) * NUV + col;
            float uval = (float)uv[idx];
            uv[idx] = (__bf16)(uval * acc[nc][v]);
        }
    }
}

// ============================================================
// p=3 power pooling over T -> bf16 pooled
// ============================================================
__global__ void pool_kernel(const float* __restrict__ h, __bf16* __restrict__ pooled) {
    int idx = blockIdx.x * 256 + threadIdx.x;   // B*D
    int d = idx & (DD - 1), b = idx >> 8;
    const float* hp = h + (size_t)b * TT * DD + d;
    float acc = 0.0f;
    for (int t = 0; t < TT; ++t) {
        float v = hp[(size_t)t * DD];
        v = (v > EPSV) ? v : EPSV;
        acc += v * v * v;
    }
    pooled[idx] = (__bf16)cbrtf(acc * (1.0f / TT));
}

// ============================================================
// row-wise L2 normalize emb (B x 256)
// ============================================================
__global__ void l2norm_kernel(float* __restrict__ emb) {
    __shared__ float red[256];
    int b = blockIdx.x, tid = threadIdx.x;
    float v = emb[(size_t)b * DD + tid];
    red[tid] = v * v;
    __syncthreads();
    for (int s = 128; s > 0; s >>= 1) {
        if (tid < s) red[tid] += red[tid + s];
        __syncthreads();
    }
    emb[(size_t)b * DD + tid] = v / sqrtf(red[0]);
}

// ============================================================
// host launcher
// ============================================================
extern "C" void kernel_launch(void* const* d_in, const int* in_sizes, int n_in,
                              void* d_out, int out_size, void* d_ws, size_t ws_size,
                              hipStream_t stream) {
    const float* x          = (const float*)d_in[0];
    const float* pe_scale   = (const float*)d_in[1];
    const float* W_in       = (const float*)d_in[2];
    const float* b_in       = (const float*)d_in[3];
    const float* norm_scale = (const float*)d_in[4];
    const float* W1         = (const float*)d_in[5];
    const float* b1         = (const float*)d_in[6];
    const float* gamma      = (const float*)d_in[7];
    const float* beta       = (const float*)d_in[8];
    const float* W2         = (const float*)d_in[9];
    const float* b2         = (const float*)d_in[10];
    const float* We         = (const float*)d_in[11];
    const float* be         = (const float*)d_in[12];

    float* emb = (float*)d_out;              // B*D
    float* h   = emb + (size_t)BB * DD;      // unpooled lives directly in d_out

    // workspace carve-up (bytes, 256-aligned)
    char* ws = (char*)d_ws;
    float*  pe       = (float*)(ws + 0);                       //  49,152 B
    __bf16* W1bf     = (__bf16*)(ws + 49152);                  // 655,360 B
    __bf16* W2bf     = (__bf16*)(ws + 704512);                 // 262,144 B
    __bf16* Webf     = (__bf16*)(ws + 966656);                 // 131,072 B
    __bf16* pooledbf = (__bf16*)(ws + 1097728);                //  65,536 B
    __bf16* hn       = (__bf16*)(ws + 1163264);                //  33.5 MB
    __bf16* uv       = (__bf16*)(ws + 34717696ULL);            //  83.9 MB
    __bf16* qb       = (__bf16*)(ws + 118603776ULL);           //  16.8 MB
    __bf16* kTb      = (__bf16*)(ws + 135380992ULL);           //  16.8 MB

    // weights -> bf16
    f2bf_kernel<<<(LL*DD*NUV + 255) / 256, 256, 0, stream>>>(W1, W1bf, LL*DD*NUV);
    f2bf_kernel<<<(LL*DD*DD  + 255) / 256, 256, 0, stream>>>(W2, W2bf, LL*DD*DD);
    f2bf_kernel<<<(DD*DD     + 255) / 256, 256, 0, stream>>>(We, Webf, DD*DD);

    // positional table + embed
    pe_kernel<<<(TT*BIN + 255) / 256, 256, 0, stream>>>(pe);
    embed_kernel<<<ROWS * DD / 256, 256, 0, stream>>>(x, pe, pe_scale, W_in, b_in, h);

    for (int i = 0; i < LL; ++i) {
        rmsnorm_kernel<<<ROWS / 8, 256, 0, stream>>>(h, norm_scale, i, hn);

        // uv = swish(hn @ W1[i] + b1[i])   (M=65536, N=640, K=256) -> bf16
        gemm_wmma_kernel<<<dim3(NUV / 64, ROWS / 128), 256, 0, stream>>>(
            hn, DD, W1bf + (size_t)i * DD * NUV, NUV, b1 + i * NUV,
            nullptr, nullptr, uv, NUV, ROWS, NUV, DD, 1);

        // q,kT = rope(base*gamma+beta)
        rope_kernel<<<ROWS * SS / 256, 256, 0, stream>>>(
            uv, gamma + i * 2 * SS, beta + i * 2 * SS, qb, kTb);

        // fused squared-ReLU attention; writes y = u * (attn@v) in place
        attn_kernel<<<dim3(TT / 16 / 4, BB), 128, 0, stream>>>(qb, kTb, uv);

        // h = h + y @ W2[i] + b2[i]        (M=65536, N=256, K=256) -> fp32
        gemm_wmma_kernel<<<dim3(DD / 64, ROWS / 128), 256, 0, stream>>>(
            uv, NUV, W2bf + (size_t)i * DD * DD, DD, b2 + i * DD,
            h, h, nullptr, DD, ROWS, DD, DD, 0);
    }

    // pooled -> emb = normalize(pooled @ We + be)
    pool_kernel<<<BB * DD / 256, 256, 0, stream>>>(h, pooledbf);
    gemm_wmma_kernel<<<dim3(DD / 64, 1), 256, 0, stream>>>(
        pooledbf, DD, Webf, DD, be, nullptr, emb, nullptr, DD, BB, DD, DD, 0);
    l2norm_kernel<<<BB, 256, 0, stream>>>(emb);
}
